// GraphAttentionNetwork_47528108097937
// MI455X (gfx1250) — matrix operand order repaired and checked by reading, not compiled, not run on hardware
//
#include <hip/hip_runtime.h>

#define N_NODES 4096
#define HD 64
#define RTYPES 3
#define JSPLIT 4
#define JCHUNK (N_NODES / JSPLIT)   // 1024 columns per block
#define LOG2E 1.44269504088896340736f

typedef __attribute__((ext_vector_type(8)))  float   v8f;
typedef __attribute__((ext_vector_type(16))) __bf16  v16bf;

union AFrag { v16bf v; __bf16 b[16]; };
union BFrag { v16bf v; uint4 q[2]; };

__device__ __forceinline__ float bpermf(float v, int srcLane) {
  return __int_as_float(__builtin_amdgcn_ds_bpermute(srcLane << 2, __float_as_int(v)));
}

// ---------------- adjacency -> bitmask (read 200MB of int32 once, emit 6MB) ----
__global__ __launch_bounds__(256) void pack_masks_kernel(
    const int* __restrict__ A0, const int* __restrict__ A1,
    const int* __restrict__ A2, unsigned* __restrict__ Ap) {
  const int r = blockIdx.y;
  const int* A = (r == 0) ? A0 : ((r == 1) ? A1 : A2);
  size_t idx = (size_t)blockIdx.x * 256 + threadIdx.x;          // over N*N
  unsigned word = __builtin_amdgcn_ballot_w32(A[idx] != 0);     // wave32 ballot
  if ((threadIdx.x & 31) == 0)
    Ap[(size_t)r * ((size_t)N_NODES * N_NODES / 32) + (idx >> 5)] = word;
}

// ---------------- Wh = H @ W[r];  WhT (bf16, transposed), f_i, f_j ------------
__global__ __launch_bounds__(256) void prep_kernel(
    const float* __restrict__ H, const float* __restrict__ W,
    const float* __restrict__ a, int D,
    unsigned short* __restrict__ WhT, float* __restrict__ Fi,
    float* __restrict__ Fj) {
  __shared__ float sH[64 * 128];
  __shared__ float sW[128 * 64];
  const int r = blockIdx.y;
  const int row0 = blockIdx.x * 64;
  const float* Wr = W + (size_t)r * D * HD;

  for (int i = threadIdx.x; i < 64 * D; i += 256) sH[i] = H[(size_t)row0 * D + i];
  for (int i = threadIdx.x; i < D * HD; i += 256) sW[i] = Wr[i];
  __syncthreads();

  const int row = threadIdx.x >> 2;          // 0..63
  const int cb  = (threadIdx.x & 3) * 16;    // output-column base
  float acc[16];
#pragma unroll
  for (int c = 0; c < 16; ++c) acc[c] = 0.f;
  for (int d = 0; d < D; ++d) {
    float h = sH[row * D + d];
    const float* wrow = &sW[d * HD + cb];
#pragma unroll
    for (int c = 0; c < 16; ++c) acc[c] = fmaf(h, wrow[c], acc[c]);
  }
  // WhT[r][col][node] (bf16, transposed so B fragments are contiguous)
  const size_t base = (size_t)r * HD * N_NODES;
#pragma unroll
  for (int c = 0; c < 16; ++c) {
    __bf16 hb = (__bf16)acc[c];
    WhT[base + (size_t)(cb + c) * N_NODES + row0 + row] =
        *reinterpret_cast<unsigned short*>(&hb);
  }

  // f_i = Wh . a_l, f_j = Wh . a_r  (reduce across the 4 lanes sharing a row)
  const float* ar = a + r * (2 * HD);
  float pi = 0.f, pj = 0.f;
#pragma unroll
  for (int c = 0; c < 16; ++c) {
    pi += acc[c] * ar[cb + c];
    pj += acc[c] * ar[HD + cb + c];
  }
  const int lane = threadIdx.x & 31;
  pi += bpermf(pi, lane ^ 1); pi += bpermf(pi, lane ^ 2);
  pj += bpermf(pj, lane ^ 1); pj += bpermf(pj, lane ^ 2);
  if ((threadIdx.x & 3) == 0) {
    Fi[r * N_NODES + row0 + row] = pi;
    Fj[r * N_NODES + row0 + row] = pj;
  }
}

// ---------------- per-relation max_j f_j (for static softmax shift) -----------
__global__ __launch_bounds__(256) void fjmax_kernel(const float* __restrict__ Fj,
                                                    float* __restrict__ Fjm) {
  __shared__ float red[256];
  const int r = blockIdx.x;
  float mx = -__builtin_inff();
  for (int i = threadIdx.x; i < N_NODES; i += 256)
    mx = fmaxf(mx, Fj[r * N_NODES + i]);
  red[threadIdx.x] = mx;
  __syncthreads();
  for (int s = 128; s > 0; s >>= 1) {
    if (threadIdx.x < s) red[threadIdx.x] = fmaxf(red[threadIdx.x], red[threadIdx.x + s]);
    __syncthreads();
  }
  if (threadIdx.x == 0) Fjm[r] = red[0];
}

// ---------------- fused masked-softmax attention, partial over (r, j-range) ---
// Static-max softmax in log2 domain: inputs pre-scaled by log2(e), p = exp2(x - m).
// Row sums computed by a 5th WMMA against an all-ones B matrix (no VALU tree).
// Each block owns one (row-tile, r, jsplit) partial -> no atomics, deterministic.
__global__ __launch_bounds__(128) void gat_partial_kernel(
    const unsigned short* __restrict__ WhT, const float* __restrict__ Fi,
    const float* __restrict__ Fj, const float* __restrict__ Fjm,
    const unsigned* __restrict__ Ap, float* __restrict__ Part,
    float* __restrict__ Lpart) {
  __shared__ float sfj[JCHUNK];
  const int tid   = threadIdx.x;
  const int lane  = tid & 31;
  const int wv    = tid >> 5;
  const int r     = blockIdx.y;
  const int js    = blockIdx.z;
  const int jbase = js * JCHUNK;
  const int ibase = blockIdx.x * 64 + wv * 16;   // 16-row tile per wave
  const int mrow  = lane & 15;                   // row for A-fragment layout
  const int upper = (lane >> 4) & 1;
  const int klo   = upper * 8;                   // A-frag K base (16-bit layout)
  const int rbase = upper * 8;                   // C-tile row base
  const int coln  = lane & 15;                   // C/B column

  // stage this block's f_j slice (pre-scaled by log2e)
  for (int i = tid; i < JCHUNK; i += 128)
    sfj[i] = Fj[r * N_NODES + jbase + i] * LOG2E;
  __syncthreads();

  const float fi = Fi[r * N_NODES + ibase + mrow] * LOG2E;
  float mx = fi + Fjm[r] * LOG2E;
  mx = fmaxf(mx, 0.2f * mx);                     // row softmax shift (upper bound)
  const unsigned* aprow = Ap +
      ((size_t)r * N_NODES + ibase + mrow) * (N_NODES / 32) + (jbase >> 5);
  const unsigned short* whr = WhT + (size_t)r * HD * N_NODES + jbase;

  BFrag ones;                                    // bf16 1.0 everywhere
  ones.q[0] = make_uint4(0x3F803F80u, 0x3F803F80u, 0x3F803F80u, 0x3F803F80u);
  ones.q[1] = ones.q[0];

  const v8f vzero = {0.f, 0.f, 0.f, 0.f, 0.f, 0.f, 0.f, 0.f};
  v8f accr[4] = {vzero, vzero, vzero, vzero};
  v8f ls = vzero;                                // row-sum tile (P @ ones)

  for (int j0 = 0; j0 < JCHUNK; j0 += 32) {
    const unsigned w = aprow[j0 >> 5] >> klo;    // constant bit positions below
    AFrag af;
#pragma unroll
    for (int k = 0; k < 8; ++k) {
      float x0 = fi + sfj[j0 + klo + k];      x0 = fmaxf(x0, 0.2f * x0);  // leaky
      float x1 = fi + sfj[j0 + 16 + klo + k]; x1 = fmaxf(x1, 0.2f * x1);
      float p0 = (w & (1u << k))        ? __builtin_amdgcn_exp2f(x0 - mx) : 0.f;
      float p1 = (w & (1u << (16 + k))) ? __builtin_amdgcn_exp2f(x1 - mx) : 0.f;
      af.b[k]     = (__bf16)p0;
      af.b[8 + k] = (__bf16)p1;
    }
    // row sums: ls += P @ ones   (replicated across columns)
    ls = __builtin_amdgcn_wmma_f32_16x16x32_bf16(
        false, af.v, false, ones.v, (short)0, ls, false, false);
    // accr += P(16x32 bf16) @ Wh(32x64 bf16) via 4 WMMAs
#pragma unroll
    for (int t = 0; t < 4; ++t) {
      BFrag bf;
      const unsigned short* bp =
          whr + (size_t)(t * 16 + coln) * N_NODES + j0 + upper * 8;
      bf.q[0] = *reinterpret_cast<const uint4*>(bp);
      bf.q[1] = *reinterpret_cast<const uint4*>(bp + 16);
      accr[t] = __builtin_amdgcn_wmma_f32_16x16x32_bf16(
          false, af.v, false, bf.v, (short)0, accr[t], false, false);
    }
  }

  // write unnormalized partial tile + partial row sums (block-exclusive slabs)
  const int slab = r * JSPLIT + js;
  float* pout = Part + (size_t)slab * N_NODES * HD;
#pragma unroll
  for (int t = 0; t < 4; ++t)
#pragma unroll
    for (int v = 0; v < 8; ++v)
      pout[(size_t)(ibase + rbase + v) * HD + t * 16 + coln] = accr[t][v];

  if (coln == 0) {                               // lanes 0 & 16 hold all 16 rows
#pragma unroll
    for (int v = 0; v < 8; ++v)
      Lpart[(size_t)slab * N_NODES + ibase + rbase + v] = ls[v];
  }
}

// ---------------- combine partials: normalize, sum relations, bias, ReLU ------
__global__ __launch_bounds__(256) void finalize_kernel(
    const float* __restrict__ Part, const float* __restrict__ Lpart,
    const float* __restrict__ bias, float* __restrict__ Hout) {
  const int idx = blockIdx.x * 256 + threadIdx.x;     // over N*HD
  const int i = idx >> 6, c = idx & (HD - 1);
  float o = bias[c];
#pragma unroll
  for (int r = 0; r < RTYPES; ++r) {
    float s = 0.f, l = 0.f;
#pragma unroll
    for (int js = 0; js < JSPLIT; ++js) {
      const int slab = r * JSPLIT + js;
      s += Part[((size_t)slab * N_NODES + i) * HD + c];
      l += Lpart[(size_t)slab * N_NODES + i];
    }
    const float inv = (l > 0.f) ? 1.f / l : 0.f;      // isolated rows -> 0
    o += s * inv;
  }
  Hout[idx] = fmaxf(o, 0.f);
}

// ---------------- score head: sigmoid(H2 @ Ws + bs) ---------------------------
__global__ __launch_bounds__(256) void head_kernel(
    const float* __restrict__ H2, const float* __restrict__ Ws,
    const float* __restrict__ bs, float* __restrict__ out) {
  const int n = blockIdx.x * 256 + threadIdx.x;
  if (n >= N_NODES) return;
  float s = bs[0];
#pragma unroll
  for (int c = 0; c < HD; ++c) s += H2[(size_t)n * HD + c] * Ws[c];
  out[n] = 1.f / (1.f + __builtin_amdgcn_exp2f(-s * LOG2E));
}

extern "C" void kernel_launch(void* const* d_in, const int* in_sizes, int n_in,
                              void* d_out, int out_size, void* d_ws, size_t ws_size,
                              hipStream_t stream) {
  const float* Z  = (const float*)d_in[0];
  const int*   Ab = (const int*)d_in[1];
  const int*   Av = (const int*)d_in[2];
  const int*   Ar = (const int*)d_in[3];
  const float* W1 = (const float*)d_in[4];
  const float* a1 = (const float*)d_in[5];
  const float* b1 = (const float*)d_in[6];
  const float* W2 = (const float*)d_in[7];
  const float* a2 = (const float*)d_in[8];
  const float* b2 = (const float*)d_in[9];
  const float* Ws = (const float*)d_in[10];
  const float* bs = (const float*)d_in[11];
  float* out = (float*)d_out;

  char* ws = (char*)d_ws;
  size_t off = 0;
  auto take = [&](size_t bytes) -> void* {
    void* p = ws + off;
    off += (bytes + 255) & ~(size_t)255;
    return p;
  };
  unsigned*       Apk  = (unsigned*)take(3ull * N_NODES * (N_NODES / 32) * 4);  // 6 MB
  unsigned short* WhT  = (unsigned short*)take(3ull * HD * N_NODES * 2);        // 1.5 MB
  float*          FiB  = (float*)take(3ull * N_NODES * 4);
  float*          FjB  = (float*)take(3ull * N_NODES * 4);
  float*          FjM  = (float*)take(RTYPES * sizeof(float));
  float*          Part = (float*)take((size_t)RTYPES * JSPLIT * N_NODES * HD * 4); // 12 MB
  float*          Lpt  = (float*)take((size_t)RTYPES * JSPLIT * N_NODES * 4);      // 192 KB
  float*          H1   = (float*)take((size_t)N_NODES * HD * 4);
  float*          H2   = (float*)take((size_t)N_NODES * HD * 4);

  const dim3 gat_grid(N_NODES / 64, RTYPES, JSPLIT);

  pack_masks_kernel<<<dim3(N_NODES * N_NODES / 256, 3), 256, 0, stream>>>(Ab, Av, Ar, Apk);

  prep_kernel<<<dim3(N_NODES / 64, 3), 256, 0, stream>>>(Z, W1, a1, 128, WhT, FiB, FjB);
  fjmax_kernel<<<RTYPES, 256, 0, stream>>>(FjB, FjM);
  gat_partial_kernel<<<gat_grid, 128, 0, stream>>>(WhT, FiB, FjB, FjM, Apk, Part, Lpt);
  finalize_kernel<<<N_NODES * HD / 256, 256, 0, stream>>>(Part, Lpt, b1, H1);

  prep_kernel<<<dim3(N_NODES / 64, 3), 256, 0, stream>>>(H1, W2, a2, 64, WhT, FiB, FjB);
  fjmax_kernel<<<RTYPES, 256, 0, stream>>>(FjB, FjM);
  gat_partial_kernel<<<gat_grid, 128, 0, stream>>>(WhT, FiB, FjB, FjM, Apk, Part, Lpt);
  finalize_kernel<<<N_NODES * HD / 256, 256, 0, stream>>>(Part, Lpt, b2, H2);

  head_kernel<<<N_NODES / 256, 256, 0, stream>>>(H2, Ws, bs, out);
}
